// SpatialProcessor_89928025243998
// MI455X (gfx1250) — matrix operand order repaired and checked by reading, not compiled, not run on hardware
//
#include <hip/hip_runtime.h>
#include <hip/hip_bf16.h>

typedef __attribute__((ext_vector_type(16))) _Float16 v16h;
typedef __attribute__((ext_vector_type(8)))  _Float16 v8h;
typedef __attribute__((ext_vector_type(8)))  float    v8f;

#define NODES 1024
#define BATCH 8
#define HEADS 4
#define NEG_SLOPE 0.2f

// Load a 16-element f16 WMMA fragment as two 16-byte chunks at p and p+second_off.
// A-fragments: second_off = 8 (K {0..15} / {16..31} per lane half).
// B-fragments: second_off = 8  (16 contiguous K values per lane).
__device__ __forceinline__ v16h load_frag(const _Float16* __restrict__ p, int second_off) {
    v8h lo = *(const v8h*)p;
    v8h hi = *(const v8h*)(p + second_off);
    v16h r;
    #pragma unroll
    for (int e = 0; e < 8; ++e) { r[e] = lo[e]; r[e + 8] = hi[e]; }
    return r;
}

// ---------------- small utility kernels ----------------

__global__ void cvt_f16_kernel(const float* __restrict__ in, _Float16* __restrict__ out, int n) {
    int t = blockIdx.x * blockDim.x + threadIdx.x;
    if (t < n) out[t] = (_Float16)in[t];
}

// in: [Kr][Nc] f32 row-major  ->  out: [Nc][Kr] f16 (weight transposed for WMMA B loads)
__global__ void transpose_cvt_kernel(const float* __restrict__ in, _Float16* __restrict__ out,
                                     int Kr, int Nc) {
    int t = blockIdx.x * blockDim.x + threadIdx.x;
    if (t >= Kr * Nc) return;
    int k = t / Nc, n = t % Nc;
    out[n * Kr + k] = (_Float16)in[t];
}

// embed [N][256] -> net [256][N] with row L2 normalization (transposed for coalesced mask dots)
__global__ void normalize_kernel(const float* __restrict__ embed, float* __restrict__ net) {
    int wave = (blockIdx.x * blockDim.x + threadIdx.x) >> 5;
    int lane = threadIdx.x & 31;
    if (wave >= NODES) return;
    float s = 0.f;
    for (int k = lane; k < 256; k += 32) { float v = embed[wave * 256 + k]; s += v * v; }
    #pragma unroll
    for (int m = 16; m >= 1; m >>= 1) s += __shfl_xor(s, m, 32);
    float inv = rsqrtf(s);
    for (int k = lane; k < 256; k += 32) net[k * NODES + wave] = embed[wave * 256 + k] * inv;
}

// maskbits[i][w] bit l  = (cos(i, w*32+l) > 0.5)
__global__ void mask_kernel(const float* __restrict__ net, unsigned* __restrict__ maskbits) {
    int wave = (blockIdx.x * blockDim.x + threadIdx.x) >> 5;
    int lane = threadIdx.x & 31;
    int i = wave >> 5;          // node row
    int wg = wave & 31;         // 32-column group
    int j = wg * 32 + lane;
    float s = 0.f;
    #pragma unroll 4
    for (int k = 0; k < 256; ++k)
        s += net[k * NODES + i] * net[k * NODES + j];
    unsigned long long bal = __ballot(s > 0.5f);
    if (lane == 0) maskbits[i * 32 + wg] = (unsigned)bal;
}

// ---------------- WMMA GEMM: C[M][Nn] = A[M][K] * Bt[Nn][K]^T (+bias), f16 in, f16 out ---------
// Software-pipelined: fragments for k-tile t+1 are loaded before the WMMAs of tile t issue.
__global__ void gemm_f16_kernel(const _Float16* __restrict__ A, const _Float16* __restrict__ Bt,
                                const float* __restrict__ bias, _Float16* __restrict__ Co,
                                int M, int Nn, int K) {
    int wave = (blockIdx.x * blockDim.x + threadIdx.x) >> 5;
    int lane = threadIdx.x & 31;
    int ntiles = Nn >> 6;                    // 64-wide N super-tiles
    int mt = wave / ntiles, nt = wave % ntiles;
    if (mt * 16 >= M) return;
    int m0 = mt * 16, n0 = nt * 64;
    int lhalf = lane >> 4, l15 = lane & 15;
    const _Float16* arow = A + (size_t)(m0 + l15) * K + lhalf * 16;
    const _Float16* brow[4];
    #pragma unroll
    for (int c = 0; c < 4; ++c)
        brow[c] = Bt + (size_t)(n0 + c * 16 + l15) * K + lhalf * 16;

    v8f acc[4] = {};
    v16h av = load_frag(arow, 8);
    v16h bv[4];
    #pragma unroll
    for (int c = 0; c < 4; ++c) bv[c] = load_frag(brow[c], 8);

    int kk = 0;
    for (; kk + 32 < K; kk += 32) {
        // issue next tile's loads before consuming the current fragments
        v16h av2 = load_frag(arow + kk + 32, 8);
        v16h bv2[4];
        #pragma unroll
        for (int c = 0; c < 4; ++c) bv2[c] = load_frag(brow[c] + kk + 32, 8);
        __builtin_prefetch(arow + kk + 64, 0, 1);     // global_prefetch_b8, 2 tiles ahead
        #pragma unroll
        for (int c = 0; c < 4; ++c)
            acc[c] = __builtin_amdgcn_wmma_f32_16x16x32_f16(
                false, av, false, bv[c], (short)0, acc[c], false, false);
        av = av2;
        #pragma unroll
        for (int c = 0; c < 4; ++c) bv[c] = bv2[c];
    }
    #pragma unroll
    for (int c = 0; c < 4; ++c)
        acc[c] = __builtin_amdgcn_wmma_f32_16x16x32_f16(
            false, av, false, bv[c], (short)0, acc[c], false, false);

    #pragma unroll
    for (int c = 0; c < 4; ++c) {
        #pragma unroll
        for (int r = 0; r < 8; ++r) {
            int row = m0 + r + lhalf * 8;
            int col = n0 + c * 16 + l15;
            float v = acc[c][r];
            if (bias) v += bias[col];
            Co[(size_t)row * Nn + col] = (_Float16)v;
        }
    }
}

// asrc/adst[b,h,n] = sum_d g[b,n,h*D+d] * a_{src,dst}[h,d]
__global__ void aproj_kernel(const _Float16* __restrict__ g, const float* __restrict__ a_src,
                             const float* __restrict__ a_dst, float* __restrict__ asrc,
                             float* __restrict__ adst, int D) {
    int t = blockIdx.x * blockDim.x + threadIdx.x;    // ((b*H+h)<<10)+n
    if (t >= BATCH * HEADS * NODES) return;
    int n = t & (NODES - 1);
    int bh = t >> 10;
    int h = bh & (HEADS - 1);
    int b = bh >> 2;
    const _Float16* gp = g + (size_t)(b * NODES + n) * (HEADS * D) + h * D;
    float s1 = 0.f, s2 = 0.f;
    for (int d = 0; d < D; ++d) {
        float v = (float)gp[d];
        s1 += v * a_src[h * D + d];
        s2 += v * a_dst[h * D + d];
    }
    asrc[t] = s1;
    adst[t] = s2;
}

// per (b,h,i): rmax = lrelu(adst_i + max_valid asrc_j); rinv = 1/sum_valid exp(lrelu(..)-rmax)
__global__ void stats_kernel(const float* __restrict__ asrc, const float* __restrict__ adst,
                             const unsigned* __restrict__ maskbits,
                             float* __restrict__ rmax, float* __restrict__ rinv) {
    int wave = (blockIdx.x * blockDim.x + threadIdx.x) >> 5;
    int lane = threadIdx.x & 31;
    if (wave >= BATCH * HEADS * NODES) return;
    int i = wave & (NODES - 1);
    int bh = wave >> 10;
    const float* as = asrc + (bh << 10);
    float adsti = adst[wave];
    float vmax = -1e30f;
    for (int w = 0; w < 32; ++w) {
        unsigned word = maskbits[i * 32 + w];
        float a = as[w * 32 + lane];
        if ((word >> lane) & 1u) vmax = fmaxf(vmax, a);
    }
    #pragma unroll
    for (int m = 16; m >= 1; m >>= 1) vmax = fmaxf(vmax, __shfl_xor(vmax, m, 32));
    float e0 = adsti + vmax;
    float me = e0 > 0.f ? e0 : NEG_SLOPE * e0;
    float s = 0.f;
    for (int w = 0; w < 32; ++w) {
        unsigned word = maskbits[i * 32 + w];
        float a = as[w * 32 + lane];
        float ev = adsti + a;
        ev = ev > 0.f ? ev : NEG_SLOPE * ev;
        if ((word >> lane) & 1u) s += __expf(ev - me);
    }
    #pragma unroll
    for (int m = 16; m >= 1; m >>= 1) s += __shfl_xor(s, m, 32);
    if (lane == 0) { rmax[wave] = me; rinv[wave] = 1.f / s; }
}

// g[b,n,h*D+d] -> gt[b,h,d,n] so aggregation B-fragments are contiguous along j
__global__ void transpose_g_kernel(const _Float16* __restrict__ g, _Float16* __restrict__ gt, int D) {
    int t = blockIdx.x * blockDim.x + threadIdx.x;
    int HD = HEADS * D;
    if (t >= BATCH * NODES * HD) return;
    int hd = t % HD;
    int n  = (t / HD) & (NODES - 1);
    int b  = t / (HD * NODES);
    int h = hd / D, d = hd % D;
    gt[(((size_t)(b * HEADS + h) * D + d) << 10) + n] = g[t];
}

// ---------------- attention aggregation: out = softmax(e) @ h, via WMMA -----------------------
// B tiles double-buffered across j-tiles so the exp-heavy A build overlaps the next loads.
template <int D, bool RELU, bool FINAL>
__global__ void agg_kernel(const _Float16* __restrict__ gt, const float* __restrict__ asrc,
                           const float* __restrict__ adst, const float* __restrict__ rmax,
                           const float* __restrict__ rinv, const unsigned* __restrict__ maskbits,
                           const float* __restrict__ bias, _Float16* __restrict__ outh,
                           float* __restrict__ outf) {
    constexpr int NDT = D / 16;
    int wave = (blockIdx.x * blockDim.x + threadIdx.x) >> 5;
    int lane = threadIdx.x & 31;
    if (wave >= BATCH * HEADS * (NODES / 16)) return;
    int it = wave & 63;
    int bh = wave >> 6;
    int h = bh & (HEADS - 1);
    int b = bh >> 2;
    int i0 = it * 16;
    int lhalf = lane >> 4, l15 = lane & 15;
    int iA = i0 + l15;                               // A-fragment row for this lane
    float adsti = adst[(bh << 10) + iA];
    float mei   = rmax[(bh << 10) + iA];
    const float* as = asrc + (bh << 10);
    const unsigned* mrow = maskbits + iA * 32;
    int kA = lhalf * 16;
    const _Float16* brow[NDT];
    #pragma unroll
    for (int c = 0; c < NDT; ++c)
        brow[c] = gt + (((size_t)bh * D + c * 16 + l15) << 10) + lhalf * 16;

    v8f acc[NDT] = {};
    v16h bv[NDT];
    #pragma unroll
    for (int c = 0; c < NDT; ++c) bv[c] = load_frag(brow[c], 8);

    for (int jt = 0; jt < NODES / 32; ++jt) {
        int j0 = jt * 32;
        unsigned word = mrow[jt];
        // prefetch next j-tile's B fragments before the exp-heavy A build
        v16h bv2[NDT];
        if (jt + 1 < NODES / 32) {
            #pragma unroll
            for (int c = 0; c < NDT; ++c) bv2[c] = load_frag(brow[c] + j0 + 32, 8);
        }
        v16h av;
        #pragma unroll
        for (int e = 0; e < 16; ++e) {
            int K = kA + e;    // {0..15} or {16..31}
            float a = as[j0 + K];
            float ev = adsti + a;
            ev = ev > 0.f ? ev : NEG_SLOPE * ev;
            float w = ((word >> K) & 1u) ? __expf(ev - mei) : 0.f;
            av[e] = (_Float16)w;
        }
        #pragma unroll
        for (int c = 0; c < NDT; ++c)
            acc[c] = __builtin_amdgcn_wmma_f32_16x16x32_f16(
                false, av, false, bv[c], (short)0, acc[c], false, false);
        #pragma unroll
        for (int c = 0; c < NDT; ++c) bv[c] = bv2[c];
    }
    #pragma unroll
    for (int r = 0; r < 8; ++r) {
        int i = i0 + r + lhalf * 8;
        float rv = rinv[(bh << 10) + i];
        #pragma unroll
        for (int c = 0; c < NDT; ++c) {
            int d = c * 16 + l15;
            float v = acc[c][r] * rv + bias[h * D + d];
            if (RELU) v = fmaxf(v, 0.f);
            size_t oidx = (size_t)(b * NODES + i) * (HEADS * D) + h * D + d;
            if (FINAL) outf[oidx] = v;
            else       outh[oidx] = (_Float16)v;
        }
    }
}

// ---------------- host orchestration ----------------

extern "C" void kernel_launch(void* const* d_in, const int* in_sizes, int n_in,
                              void* d_out, int out_size, void* d_ws, size_t ws_size,
                              hipStream_t stream) {
    const float* x      = (const float*)d_in[0];   // [8,1024,64]
    const float* embed  = (const float*)d_in[1];   // [1024,256]
    const float* Wp     = (const float*)d_in[2];   // [64,256]
    const float* bp     = (const float*)d_in[3];   // [256]
    const float* W1     = (const float*)d_in[4];   // [256,256]
    const float* a_src1 = (const float*)d_in[5];   // [4,64]
    const float* a_dst1 = (const float*)d_in[6];
    const float* b1     = (const float*)d_in[7];   // [256]
    const float* W2     = (const float*)d_in[8];   // [256,128]
    const float* a_src2 = (const float*)d_in[9];   // [4,32]
    const float* a_dst2 = (const float*)d_in[10];
    const float* b2     = (const float*)d_in[11];  // [128]
    float* out = (float*)d_out;                    // [8,1024,128]

    char* ws = (char*)d_ws;
    size_t off = 0;
    auto alloc = [&](size_t bytes) -> char* {
        char* p = ws + off;
        off += (bytes + 255) & ~(size_t)255;
        return p;
    };
    const int BN = BATCH * NODES;                          // 8192
    _Float16* xh   = (_Float16*)alloc((size_t)BN * 64 * 2);
    _Float16* WpT  = (_Float16*)alloc(256 * 64 * 2);
    _Float16* W1T  = (_Float16*)alloc(256 * 256 * 2);
    _Float16* W2T  = (_Float16*)alloc(128 * 256 * 2);
    float*    net  = (float*)   alloc(256 * NODES * 4);
    unsigned* mb   = (unsigned*)alloc(NODES * 32 * 4);
    _Float16* h0   = (_Float16*)alloc((size_t)BN * 256 * 2);
    _Float16* g    = (_Float16*)alloc((size_t)BN * 256 * 2);   // reused for layer2 (8192x128)
    _Float16* gt   = (_Float16*)alloc((size_t)BN * 256 * 2);   // reused for layer2
    _Float16* h1   = (_Float16*)alloc((size_t)BN * 256 * 2);
    float* asrc = (float*)alloc(BATCH * HEADS * NODES * 4);
    float* adst = (float*)alloc(BATCH * HEADS * NODES * 4);
    float* rmx  = (float*)alloc(BATCH * HEADS * NODES * 4);
    float* rnv  = (float*)alloc(BATCH * HEADS * NODES * 4);
    (void)ws_size; (void)in_sizes; (void)n_in; (void)out_size;

    // input conversions
    cvt_f16_kernel<<<(BN * 64 + 255) / 256, 256, 0, stream>>>(x, xh, BN * 64);
    transpose_cvt_kernel<<<(64 * 256 + 255) / 256, 256, 0, stream>>>(Wp, WpT, 64, 256);
    transpose_cvt_kernel<<<(256 * 256 + 255) / 256, 256, 0, stream>>>(W1, W1T, 256, 256);
    transpose_cvt_kernel<<<(256 * 128 + 255) / 256, 256, 0, stream>>>(W2, W2T, 256, 128);

    // adjacency mask
    normalize_kernel<<<NODES / 8, 256, 0, stream>>>(embed, net);
    mask_kernel<<<(NODES * 32) / 8, 256, 0, stream>>>(net, mb);

    // h0 = x @ Wp + bp   (8192x64 @ 64x256)
    gemm_f16_kernel<<<(BN / 16) * (256 / 64) / 8, 256, 0, stream>>>(xh, WpT, bp, h0, BN, 256, 64);

    // -------- GAT layer 1 --------
    gemm_f16_kernel<<<(BN / 16) * (256 / 64) / 8, 256, 0, stream>>>(h0, W1T, nullptr, g, BN, 256, 256);
    aproj_kernel<<<(BATCH * HEADS * NODES) / 256, 256, 0, stream>>>(g, a_src1, a_dst1, asrc, adst, 64);
    stats_kernel<<<(BATCH * HEADS * NODES) / 8, 256, 0, stream>>>(asrc, adst, mb, rmx, rnv);
    transpose_g_kernel<<<(BN * 256) / 256, 256, 0, stream>>>(g, gt, 64);
    agg_kernel<64, true, false><<<(BATCH * HEADS * (NODES / 16)) / 8, 256, 0, stream>>>(
        gt, asrc, adst, rmx, rnv, mb, b1, h1, nullptr);

    // -------- GAT layer 2 --------
    gemm_f16_kernel<<<(BN / 16) * (128 / 64) / 8, 256, 0, stream>>>(h1, W2T, nullptr, g, BN, 128, 256);
    aproj_kernel<<<(BATCH * HEADS * NODES) / 256, 256, 0, stream>>>(g, a_src2, a_dst2, asrc, adst, 32);
    stats_kernel<<<(BATCH * HEADS * NODES) / 8, 256, 0, stream>>>(asrc, adst, mb, rmx, rnv);
    transpose_g_kernel<<<(BN * 128) / 256, 256, 0, stream>>>(g, gt, 32);
    agg_kernel<32, false, true><<<(BATCH * HEADS * (NODES / 16)) / 8, 256, 0, stream>>>(
        gt, asrc, adst, rmx, rnv, mb, b2, nullptr, out);
}
